// GraphSSM_43138651521082
// MI455X (gfx1250) — matrix operand order repaired and checked by reading, not compiled, not run on hardware
//
#include <hip/hip_runtime.h>
#include <hip/hip_bf16.h>

#define L_SEQ   2048
#define DMODEL  512
#define DINNER  1024
#define NSTATE  16
#define DTRANK  32

// LDS B-tile geometry: 64 rows (output cols) x 32 k-halfs, padded to 40 halfs/row
#define BROW    40
#define BTILE   (64 * BROW)

typedef _Float16 h8   __attribute__((ext_vector_type(8)));
typedef _Float16 v16h __attribute__((ext_vector_type(16)));
typedef float    v8f  __attribute__((ext_vector_type(8)));

union HU { v16h v; h8 h[2]; };

// ---- WMMA fragment helpers (A/B in contiguous k order, wave32) ----------

// A fragment: 16x32 f16, row-major source, leading dim ld.
// lane<16 holds row M=lane, K in {0..15}; lane>=16: K in {16..31}.
__device__ __forceinline__ v16h load_frag_a(const _Float16* __restrict__ src, int ld,
                                            int row_base, int k_base, int lane) {
  int m  = lane & 15;
  int kh = (lane >> 4) << 4;                       // 0 or 16
  const _Float16* p = src + (size_t)(row_base + m) * ld + (k_base + kh);
  HU u;
  u.h[0] = *reinterpret_cast<const h8*>(p);
  u.h[1] = *reinterpret_cast<const h8*>(p + 8);
  return u.v;
}

// B fragment from an LDS-staged tile: B[k][n] = W[n][k]; tile row = output col,
// BROW-half pitch. lane n holds column n; lane-half selects K 0..15 / 16..31.
__device__ __forceinline__ v16h frag_b_lds(const _Float16* sbuf, int jrow16, int lane) {
  int n  = lane & 15;
  int kl = (lane >> 4) << 4;                       // 0 or 16
  const _Float16* p = sbuf + (jrow16 + n) * BROW + kl;
  HU u;
  u.h[0] = *reinterpret_cast<const h8*>(p);        // ds_load_b128
  u.h[1] = *reinterpret_cast<const h8*>(p + 8);
  return u.v;
}

// B fragment straight from row-major global W (used by the K=32 dt_proj GEMM).
__device__ __forceinline__ v16h load_frag_b(const _Float16* __restrict__ w, int ld,
                                            int col_base, int k_base, int lane) {
  int n  = lane & 15;
  int kl = (lane >> 4) << 4;
  const _Float16* p = w + (size_t)(col_base + n) * ld + (k_base + kl);
  HU u;
  u.h[0] = *reinterpret_cast<const h8*>(p);
  u.h[1] = *reinterpret_cast<const h8*>(p + 8);
  return u.v;
}

__device__ __forceinline__ v8f wmma16(v16h a, v16h b, v8f c) {
  return __builtin_amdgcn_wmma_f32_16x16x32_f16(false, a, false, b, (short)0, c,
                                                false, false);
}

// C/D store, row-major: VGPR r -> M = row_base + 8*(lane>>4) + r, N = col_base + (lane&15)
__device__ __forceinline__ void store_frag(float* __restrict__ C, int ld,
                                           int row_base, int col_base, int lane, v8f acc) {
  int n  = lane & 15;
  int mo = (lane >> 4) << 3;
  #pragma unroll
  for (int r = 0; r < 8; ++r)
    C[(size_t)(row_base + mo + r) * ld + col_base + n] = acc[r];
}

// Transposed store: element (M,N) -> CT[N][M]
__device__ __forceinline__ void store_frag_T(float* __restrict__ CT, int ldT,
                                             int row_base, int col_base, int lane, v8f acc) {
  int n  = lane & 15;
  int mo = (lane >> 4) << 3;
  float* p = CT + (size_t)(col_base + n) * ldT + row_base + mo;
  #pragma unroll
  for (int r = 0; r < 8; ++r) p[r] = acc[r];
}

// ---- Async staging: one global_load_async_to_lds_b128 (16B) per thread -----
// 256 threads stage a 64x32-half B tile: thread -> row r = tid>>2, 8-half chunk c.
__device__ __forceinline__ void stage_b_async(_Float16* sbuf,
                                              const _Float16* __restrict__ W, int ldb,
                                              int col_base, int k, int tid) {
  int r = tid >> 2;
  int c = tid & 3;
  const _Float16* g = W + (size_t)(col_base + r) * ldb + k + c * 8;
  unsigned lds = (unsigned)(uintptr_t)(sbuf + r * BROW + c * 8);
  asm volatile("global_load_async_to_lds_b128 %0, %1, off"
               :: "v"(lds), "v"(g) : "memory");
}

__device__ __forceinline__ void wait_async0() {
  asm volatile("s_wait_asynccnt 0x0" ::: "memory");
}

// ---- Shared GEMM inner loop: M16(wave) x N64 tile, LDS-staged ping-pong B --
template<int KDIM>
__device__ __forceinline__ void gemm_loop(const _Float16* __restrict__ A, int lda,
                                          const _Float16* __restrict__ W, int ldb,
                                          int row_base, int col_base,
                                          _Float16* sB, v8f acc[4]) {
  int tid  = threadIdx.x;
  int lane = tid & 31;
  stage_b_async(sB, W, ldb, col_base, 0, tid);
  wait_async0();
  __syncthreads();
  v16h a = load_frag_a(A, lda, row_base, 0, lane);
  for (int k = 0; k < KDIM; k += 32) {
    _Float16* bcur = sB + ((k >> 5) & 1) * BTILE;
    v16h a_next = a;
    if (k + 32 < KDIM) {
      stage_b_async(sB + (((k >> 5) & 1) ^ 1) * BTILE, W, ldb, col_base, k + 32, tid);
      a_next = load_frag_a(A, lda, row_base, k + 32, lane);
    }
    v16h b0 = frag_b_lds(bcur, 0,  lane);
    v16h b1 = frag_b_lds(bcur, 16, lane);
    v16h b2 = frag_b_lds(bcur, 32, lane);
    v16h b3 = frag_b_lds(bcur, 48, lane);
    acc[0] = wmma16(a, b0, acc[0]);
    acc[1] = wmma16(a, b1, acc[1]);
    acc[2] = wmma16(a, b2, acc[2]);
    acc[3] = wmma16(a, b3, acc[3]);
    a = a_next;
    if (k + 32 < KDIM) {
      wait_async0();
      __syncthreads();
    }
  }
}

// ---- Kernel 0: f32 -> f16 cast ---------------------------------------------
__global__ void k_cvt(const float* __restrict__ src, _Float16* __restrict__ dst, int n) {
  int i = blockIdx.x * 256 + threadIdx.x;
  if (i < n) dst[i] = (_Float16)src[i];
}

// ---- Kernel 1: in_proj GEMM, split + transposed stores ---------------------
__global__ void __launch_bounds__(256) k_inproj(const _Float16* __restrict__ X16,
                                                const _Float16* __restrict__ W16,
                                                float* __restrict__ hiddenT,
                                                float* __restrict__ gateT) {
  __shared__ _Float16 sB[2 * BTILE];
  int lane = threadIdx.x & 31;
  int wave = threadIdx.x >> 5;
  int row_base = blockIdx.y * 128 + wave * 16;
  int col_base = blockIdx.x * 64;
  v8f z = {0.f,0.f,0.f,0.f,0.f,0.f,0.f,0.f};
  v8f acc[4] = {z, z, z, z};
  gemm_loop<DMODEL>(X16, DMODEL, W16, DMODEL, row_base, col_base, sB, acc);
  float* dstT = (col_base < DINNER) ? hiddenT : gateT;
  int cb = (col_base < DINNER) ? col_base : col_base - DINNER;
  #pragma unroll
  for (int j = 0; j < 4; ++j)
    store_frag_T(dstT, L_SEQ, row_base, cb + 16 * j, lane, acc[j]);
}

// ---- Kernel 2: depthwise conv(4) + bias + SiLU -----------------------------
__global__ void __launch_bounds__(256) k_conv(const float* __restrict__ hiddenT,
                                              const float* __restrict__ cw,
                                              const float* __restrict__ cb,
                                              float* __restrict__ convT,
                                              _Float16* __restrict__ conv16) {
  int gid = blockIdx.x * 256 + threadIdx.x;
  int l = gid & (L_SEQ - 1);
  int d = gid >> 11;
  const float* xr = hiddenT + (size_t)d * L_SEQ;
  float s = cb[d];
  #pragma unroll
  for (int j = 0; j < 4; ++j) {
    int ll = l - 3 + j;
    float x = (ll >= 0) ? xr[ll] : 0.f;
    s += x * cw[d * 4 + j];
  }
  float v = s / (1.f + expf(-s));               // SiLU
  convT[(size_t)d * L_SEQ + l] = v;
  conv16[(size_t)l * DINNER + d] = (_Float16)v;
}

// ---- Kernel 3: x_proj GEMM -> ssm[L,64] (dt|B|C), dt part also f16 ---------
__global__ void __launch_bounds__(256) k_xproj(const _Float16* __restrict__ C16,
                                               const _Float16* __restrict__ XW16,
                                               float* __restrict__ ssm,
                                               _Float16* __restrict__ dt16) {
  __shared__ _Float16 sB[2 * BTILE];
  int lane = threadIdx.x & 31;
  int wave = threadIdx.x >> 5;
  int row_base = blockIdx.x * 128 + wave * 16;
  v8f z = {0.f,0.f,0.f,0.f,0.f,0.f,0.f,0.f};
  v8f acc[4] = {z, z, z, z};
  gemm_loop<DINNER>(C16, DINNER, XW16, DINNER, row_base, 0, sB, acc);
  #pragma unroll
  for (int j = 0; j < 4; ++j) store_frag(ssm, 64, row_base, 16 * j, lane, acc[j]);
  int n = lane & 15, mo = (lane >> 4) << 3;
  #pragma unroll
  for (int j = 0; j < 2; ++j)
    #pragma unroll
    for (int r = 0; r < 8; ++r)
      dt16[(size_t)(row_base + mo + r) * DTRANK + 16 * j + n] = (_Float16)acc[j][r];
}

// ---- Kernel 4: dt_proj GEMM (K=32, single WMMA step) + bias + softplus -----
__global__ void __launch_bounds__(256) k_dtproj(const _Float16* __restrict__ dt16,
                                                const _Float16* __restrict__ DTW16,
                                                const float* __restrict__ dt_b,
                                                float* __restrict__ dtT) {
  int lane = threadIdx.x & 31;
  int wave = threadIdx.x >> 5;
  int row_base = blockIdx.y * 128 + wave * 16;
  int col_base = blockIdx.x * 64;
  v8f z = {0.f,0.f,0.f,0.f,0.f,0.f,0.f,0.f};
  v16h a = load_frag_a(dt16, DTRANK, row_base, 0, lane);
  int n = lane & 15, mo = (lane >> 4) << 3;
  #pragma unroll
  for (int j = 0; j < 4; ++j) {
    v16h b = load_frag_b(DTW16, DTRANK, col_base + 16 * j, 0, lane);
    v8f acc = wmma16(a, b, z);
    int col = col_base + 16 * j + n;
    float bias = dt_b[col];
    #pragma unroll
    for (int r = 0; r < 8; ++r) {
      float v = acc[r] + bias;
      v = (v > 20.f) ? v : log1pf(expf(v));     // softplus
      dtT[(size_t)col * L_SEQ + (row_base + mo + r)] = v;
    }
  }
}

// ---- Kernel 5: selective scan (anti-causal per the reference's reverse/roll)
__global__ void __launch_bounds__(256) k_scan(const float* __restrict__ convT,
                                              const float* __restrict__ dtT,
                                              const float* __restrict__ ssm,
                                              const float* __restrict__ gateT,
                                              const float* __restrict__ A_log,
                                              const float* __restrict__ Dp,
                                              _Float16* __restrict__ out16) {
  int tid = threadIdx.x;
  int d = blockIdx.x * 16 + (tid >> 4);
  int n = tid & 15;
  float a  = -expf(A_log[d * NSTATE + n]);
  float Dd = Dp[d];
  float g = 0.f, dA_next = 0.f;                  // dA[L] == 0 -> g[L-1] = dBu[L-1]
  for (int t = L_SEQ - 1; t >= 0; --t) {
    float u  = convT[(size_t)d * L_SEQ + t];
    float dt = dtT [(size_t)d * L_SEQ + t];
    float Bn = ssm[(size_t)t * 64 + DTRANK + n];
    float Cn = ssm[(size_t)t * 64 + DTRANK + NSTATE + n];
    g = dA_next * g + dt * Bn * u;
    float y = 1.3f * g * Cn;
    y += __shfl_xor(y, 1, 32);
    y += __shfl_xor(y, 2, 32);
    y += __shfl_xor(y, 4, 32);
    y += __shfl_xor(y, 8, 32);                   // sum over 16 states
    if (n == 0) {
      float gt  = gateT[(size_t)d * L_SEQ + t];
      float val = (y + Dd * u) * (gt / (1.f + expf(-gt)));
      out16[(size_t)t * DINNER + d] = (_Float16)val;
    }
    dA_next = expf(a * dt);
  }
}

// ---- Kernel 6: out_proj GEMM -> d_out [L, 512] f32 -------------------------
__global__ void __launch_bounds__(256) k_outproj(const _Float16* __restrict__ S16,
                                                 const _Float16* __restrict__ OW16,
                                                 float* __restrict__ out) {
  __shared__ _Float16 sB[2 * BTILE];
  int lane = threadIdx.x & 31;
  int wave = threadIdx.x >> 5;
  int row_base = blockIdx.y * 128 + wave * 16;
  int col_base = blockIdx.x * 64;
  v8f z = {0.f,0.f,0.f,0.f,0.f,0.f,0.f,0.f};
  v8f acc[4] = {z, z, z, z};
  gemm_loop<DINNER>(S16, DINNER, OW16, DINNER, row_base, col_base, sB, acc);
  #pragma unroll
  for (int j = 0; j < 4; ++j)
    store_frag(out, DMODEL, row_base, col_base + 16 * j, lane, acc[j]);
}

// ---- Host launcher ---------------------------------------------------------
extern "C" void kernel_launch(void* const* d_in, const int* in_sizes, int n_in,
                              void* d_out, int out_size, void* d_ws, size_t ws_size,
                              hipStream_t stream) {
  const float* X    = (const float*)d_in[0];   // (1, 2048, 512)
  const float* Win  = (const float*)d_in[1];   // (2048, 512)
  const float* cw   = (const float*)d_in[2];   // (1024, 1, 4)
  const float* cb   = (const float*)d_in[3];   // (1024,)
  const float* xw   = (const float*)d_in[4];   // (64, 1024)
  const float* dtw  = (const float*)d_in[5];   // (1024, 32)
  const float* dtb  = (const float*)d_in[6];   // (1024,)
  const float* Alog = (const float*)d_in[7];   // (1024, 16)
  const float* Dp   = (const float*)d_in[8];   // (1024,)
  const float* ow   = (const float*)d_in[9];   // (512, 1024)
  float* out = (float*)d_out;                  // (1, 2048, 512)

  char* w = (char*)d_ws;
  auto alloc = [&](size_t bytes) { char* p = w; w += (bytes + 255) & ~(size_t)255; return p; };

  _Float16* X16    = (_Float16*)alloc((size_t)L_SEQ * DMODEL * 2);
  _Float16* Win16  = (_Float16*)alloc((size_t)2 * DINNER * DMODEL * 2);
  _Float16* xw16   = (_Float16*)alloc((size_t)64 * DINNER * 2);
  _Float16* dtw16  = (_Float16*)alloc((size_t)DINNER * DTRANK * 2);
  _Float16* ow16   = (_Float16*)alloc((size_t)DMODEL * DINNER * 2);
  float*    hidT   = (float*)   alloc((size_t)DINNER * L_SEQ * 4);
  float*    gatT   = (float*)   alloc((size_t)DINNER * L_SEQ * 4);
  float*    convT  = (float*)   alloc((size_t)DINNER * L_SEQ * 4);
  _Float16* conv16 = (_Float16*)alloc((size_t)L_SEQ * DINNER * 2);
  float*    ssm    = (float*)   alloc((size_t)L_SEQ * 64 * 4);
  _Float16* dt16   = (_Float16*)alloc((size_t)L_SEQ * DTRANK * 2);
  float*    dtT    = (float*)   alloc((size_t)DINNER * L_SEQ * 4);
  _Float16* scan16 = (_Float16*)alloc((size_t)L_SEQ * DINNER * 2);

  auto cvt = [&](const float* s, _Float16* d, int n) {
    k_cvt<<<(n + 255) / 256, 256, 0, stream>>>(s, d, n);
  };
  cvt(X,   X16,   L_SEQ * DMODEL);
  cvt(Win, Win16, 2 * DINNER * DMODEL);
  cvt(xw,  xw16,  64 * DINNER);
  cvt(dtw, dtw16, DINNER * DTRANK);
  cvt(ow,  ow16,  DMODEL * DINNER);

  k_inproj <<<dim3(32, 16), 256, 0, stream>>>(X16, Win16, hidT, gatT);
  k_conv   <<<(DINNER * L_SEQ) / 256, 256, 0, stream>>>(hidT, cw, cb, convT, conv16);
  k_xproj  <<<16, 256, 0, stream>>>(conv16, xw16, ssm, dt16);
  k_dtproj <<<dim3(16, 16), 256, 0, stream>>>(dt16, dtw16, dtb, dtT);
  k_scan   <<<DINNER / 16, 256, 0, stream>>>(convT, dtT, ssm, gatT, Alog, Dp, scan16);
  k_outproj<<<dim3(8, 16), 256, 0, stream>>>(scan16, ow16, out);
}
